// FourierLayer_11957188952745
// MI455X (gfx1250) — compile-verified, matrix operand order checked by reading, not run on hardware
//
#include <hip/hip_runtime.h>
#include <hip/hip_bf16.h>

// ---------------------------------------------------------------------------
// FourierLayer == circular convolution == GEMM against a circulant matrix.
//   out[b][i] = relu( lw[i] * ( sum_k x[b][k]*hr[(i-k)&4095] + x[b][i] ) )
//   hr[n] = (1/N) * sum_k fw[k] * cos(2*pi*k*n/N)
// GEMM: M=16384, N=4096, K=4096, bf16 WMMA with f32 accumulation.
// ---------------------------------------------------------------------------

#define BATCH 16384
#define NMODE 4096

typedef __attribute__((ext_vector_type(16))) __bf16 v16bf;
typedef __attribute__((ext_vector_type(8)))  __bf16 v8bf;
typedef __attribute__((ext_vector_type(8)))  float  v8f;

// 16-byte payload type matching the async-LDS builtin's parameter type.
typedef int v4i_pay __attribute__((vector_size(16)));
typedef __attribute__((address_space(1))) v4i_pay* gbl_v4i_ptr;
typedef __attribute__((address_space(3))) v4i_pay* lds_v4i_ptr;

// CDNA5 async global->LDS path (ASYNCcnt-tracked), if this toolchain has it.
#if defined(__has_builtin)
#  if __has_builtin(__builtin_amdgcn_global_load_async_to_lds_b128)
#    define HAVE_ASYNC_LDS 1
#  endif
#endif
#ifndef HAVE_ASYNC_LDS
#  define HAVE_ASYNC_LDS 0
#endif

__device__ __forceinline__ void async_cp_b128(const unsigned short* gsrc, unsigned short* ldst) {
#if HAVE_ASYNC_LDS
    __builtin_amdgcn_global_load_async_to_lds_b128(
        (gbl_v4i_ptr)(const void*)gsrc, (lds_v4i_ptr)(void*)ldst, 0, 0);
#else
    *reinterpret_cast<uint4*>(ldst) = *reinterpret_cast<const uint4*>(gsrc);
#endif
}

__device__ __forceinline__ void wait_async_lds() {
#if HAVE_ASYNC_LDS
#  if __has_builtin(__builtin_amdgcn_s_wait_asynccnt)
    __builtin_amdgcn_s_wait_asynccnt(0);
#  else
    asm volatile("s_wait_asynccnt 0" ::: "memory");
#  endif
#endif
}

__device__ __forceinline__ unsigned short f2bf(float f) {
    unsigned u = __float_as_uint(f);
    unsigned r = u + 0x7FFFu + ((u >> 16) & 1u);   // round-to-nearest-even
    return (unsigned short)(r >> 16);
}

// ---------------------------------------------------------------------------
// Kernel 1: hr[n] = (1/N) sum_k fw[k]*cos(2*pi*((k*n) mod N)/N)
// Integer phase reduction keeps the cosf argument in [0, 2*pi).
// ---------------------------------------------------------------------------
__global__ void build_hr(const float* __restrict__ fw, float* __restrict__ hr) {
    const int n = blockIdx.x * blockDim.x + threadIdx.x;   // 0..4095
    const float c = 6.28318530717958647692f / (float)NMODE;
    float s = 0.0f;
    for (int k = 0; k < NMODE; ++k) {
        int ph = (k * n) & (NMODE - 1);
        s += fw[k] * cosf((float)ph * c);
    }
    hr[n] = s * (1.0f / (float)NMODE);
}

// ---------------------------------------------------------------------------
// Kernel 2: circulant (stored transposed, K-contiguous): Ct[i][k] = hr[(i-k)&4095]
// ---------------------------------------------------------------------------
__global__ void build_circ(const float* __restrict__ hr, unsigned short* __restrict__ Ct) {
    const size_t e = ((size_t)blockIdx.x * blockDim.x + threadIdx.x) * 8;  // flat elem idx
    const int i = (int)(e >> 12);
    const int k = (int)(e & (NMODE - 1));
    unsigned short t[8];
#pragma unroll
    for (int j = 0; j < 8; ++j)
        t[j] = f2bf(hr[(i - (k + j)) & (NMODE - 1)]);
    uint4 u;
    u.x = (unsigned)t[0] | ((unsigned)t[1] << 16);
    u.y = (unsigned)t[2] | ((unsigned)t[3] << 16);
    u.z = (unsigned)t[4] | ((unsigned)t[5] << 16);
    u.w = (unsigned)t[6] | ((unsigned)t[7] << 16);
    *reinterpret_cast<uint4*>(Ct + e) = u;
}

// ---------------------------------------------------------------------------
// Kernel 2b: one-shot X (f32) -> Abf (bf16). Removes per-block-column
// redundant conversion from the GEMM inner loop; 128MB result fits in L2.
// ---------------------------------------------------------------------------
__global__ void convert_A(const float* __restrict__ X, unsigned short* __restrict__ Abf) {
    const size_t e = ((size_t)blockIdx.x * blockDim.x + threadIdx.x) * 8;
    const float4 f0 = *reinterpret_cast<const float4*>(X + e);
    const float4 f1 = *reinterpret_cast<const float4*>(X + e + 4);
    uint4 u;
    u.x = (unsigned)f2bf(f0.x) | ((unsigned)f2bf(f0.y) << 16);
    u.y = (unsigned)f2bf(f0.z) | ((unsigned)f2bf(f0.w) << 16);
    u.z = (unsigned)f2bf(f1.x) | ((unsigned)f2bf(f1.y) << 16);
    u.w = (unsigned)f2bf(f1.z) | ((unsigned)f2bf(f1.w) << 16);
    *reinterpret_cast<uint4*>(Abf + e) = u;
}

// ---------------------------------------------------------------------------
// Kernel 3: GEMM  O = relu(lw .* (X*C + X))
// Block: 256 threads = 8 waves (wave32). Block tile 128x128, BK=32.
// Wave grid 2(M) x 4(N): each wave 64x32 = 4x2 WMMA 16x16 accumulators.
// Double-buffered LDS. PRE=true: both tiles staged via async global->LDS
// from pre-converted bf16. PRE=false: A converted f32->bf16 while staging.
// ---------------------------------------------------------------------------
#define LDS_STRIDE 40   // 32 bf16 payload + 8 pad (80B rows, 16B aligned)

template <bool PRE>
__global__ __launch_bounds__(256)
void fourier_gemm(const float* __restrict__ X,
                  const unsigned short* __restrict__ Abf,
                  const unsigned short* __restrict__ Ct,
                  const float* __restrict__ lw,
                  float* __restrict__ out) {
    __shared__ unsigned short As[2][128][LDS_STRIDE];
    __shared__ unsigned short Bs[2][128][LDS_STRIDE];

    const int tid    = threadIdx.x;
    const int blockN = blockIdx.x * 128;
    const int blockM = blockIdx.y * 128;

    const int wave  = tid >> 5;
    const int lane  = tid & 31;
    const int waveM = wave >> 2;         // 0..1 -> 64 rows each
    const int waveN = wave & 3;          // 0..3 -> 32 cols each
    const int rl    = lane & 15;         // row/col within 16x16 tile
    const int kHalf = (lane >> 4) << 3;  // 0 or 8 (bf16 A/B fragment K layout)

    auto stage = [&](int b, int kt) {
        // B: 128 rows (output cols) x 32 bf16, K-contiguous -> async LDS copy
#pragma unroll
        for (int j = 0; j < 2; ++j) {
            int c   = tid + j * 256;
            int row = c >> 2;
            int off = (c & 3) * 8;
            async_cp_b128(Ct + (size_t)(blockN + row) * NMODE + kt * 32 + off,
                          &Bs[b][row][off]);
        }
        if (PRE) {
            // A: 128 rows x 32 bf16 -> async LDS copy (no VGPR round trip)
#pragma unroll
            for (int j = 0; j < 2; ++j) {
                int c   = tid + j * 256;
                int row = c >> 2;
                int off = (c & 3) * 8;
                async_cp_b128(Abf + (size_t)(blockM + row) * NMODE + kt * 32 + off,
                              &As[b][row][off]);
            }
        } else {
            // A: 128 rows x 32 f32 -> bf16; 1024 float4 chunks / 256 threads
#pragma unroll
            for (int j = 0; j < 4; ++j) {
                int c    = tid + j * 256;
                int row  = c >> 3;
                int coff = (c & 7) * 4;
                const float4 f = *reinterpret_cast<const float4*>(
                    X + (size_t)(blockM + row) * NMODE + kt * 32 + coff);
                ushort4 h;
                h.x = f2bf(f.x); h.y = f2bf(f.y); h.z = f2bf(f.z); h.w = f2bf(f.w);
                *reinterpret_cast<ushort4*>(&As[b][row][coff]) = h;
            }
        }
    };

    const v8f vzero = {0.f, 0.f, 0.f, 0.f, 0.f, 0.f, 0.f, 0.f};
    v8f acc[4][2];
#pragma unroll
    for (int mt = 0; mt < 4; ++mt)
#pragma unroll
        for (int nt = 0; nt < 2; ++nt) acc[mt][nt] = vzero;

    stage(0, 0);
    wait_async_lds();
    __syncthreads();

    const int KT = NMODE / 32;   // 128 k-steps
    for (int kt = 0; kt < KT; ++kt) {
        const int buf = kt & 1;
        if (kt + 1 < KT) stage(buf ^ 1, kt + 1);

        // Prefetch the A source two steps ahead into the cache hierarchy.
        if (kt + 2 < KT) {
            const char* p;
            if (PRE)
                p = reinterpret_cast<const char*>(
                    Abf + (size_t)(blockM + (tid >> 1)) * NMODE + (kt + 2) * 32 + (tid & 1) * 16);
            else
                p = reinterpret_cast<const char*>(
                    X + (size_t)(blockM + (tid >> 1)) * NMODE + (kt + 2) * 32 + (tid & 1) * 16);
            __builtin_prefetch(p, 0, 3);
        }

        // Load fragments into registers, then fire the 8 WMMAs.
        v16bf bfrag[2];
#pragma unroll
        for (int nt = 0; nt < 2; ++nt) {
            const unsigned short* p = &Bs[buf][waveN * 32 + nt * 16 + rl][0];
            v8bf lo = *reinterpret_cast<const v8bf*>(p + kHalf);
            v8bf hi = *reinterpret_cast<const v8bf*>(p + kHalf + 16);
            bfrag[nt] = __builtin_shufflevector(lo, hi,
                0, 1, 2, 3, 4, 5, 6, 7, 8, 9, 10, 11, 12, 13, 14, 15);
        }
        v16bf afrag[4];
#pragma unroll
        for (int mt = 0; mt < 4; ++mt) {
            const unsigned short* p = &As[buf][waveM * 64 + mt * 16 + rl][0];
            v8bf lo = *reinterpret_cast<const v8bf*>(p + kHalf);
            v8bf hi = *reinterpret_cast<const v8bf*>(p + kHalf + 16);
            afrag[mt] = __builtin_shufflevector(lo, hi,
                0, 1, 2, 3, 4, 5, 6, 7, 8, 9, 10, 11, 12, 13, 14, 15);
        }
#pragma unroll
        for (int mt = 0; mt < 4; ++mt)
#pragma unroll
            for (int nt = 0; nt < 2; ++nt)
                acc[mt][nt] = __builtin_amdgcn_wmma_f32_16x16x32_bf16(
                    false, afrag[mt], false, bfrag[nt], (short)0, acc[mt][nt],
                    false, false);

        if (kt + 1 < KT) wait_async_lds();
        __syncthreads();
    }

    // Epilogue: out = relu(lw * (acc + x)), skip path in f32.
    // C layout: VGPR r -> M = r + (lane>=16 ? 8 : 0); N = lane&15.
    const int mhalf = (lane >> 4) << 3;
#pragma unroll
    for (int nt = 0; nt < 2; ++nt) {
        const int col = blockN + waveN * 32 + nt * 16 + rl;
        const float w = lw[col];
#pragma unroll
        for (int mt = 0; mt < 4; ++mt) {
#pragma unroll
            for (int r = 0; r < 8; ++r) {
                const int row = blockM + waveM * 64 + mt * 16 + mhalf + r;
                const size_t idx = (size_t)row * NMODE + col;
                float v = (acc[mt][nt][r] + X[idx]) * w;
                out[idx] = v > 0.0f ? v : 0.0f;
            }
        }
    }
}

// ---------------------------------------------------------------------------
// Launch
// ---------------------------------------------------------------------------
extern "C" void kernel_launch(void* const* d_in, const int* in_sizes, int n_in,
                              void* d_out, int out_size, void* d_ws, size_t ws_size,
                              hipStream_t stream) {
    const float* X  = (const float*)d_in[0];   // [16384, 4096]
    const float* fw = (const float*)d_in[1];   // [4096]
    const float* lw = (const float*)d_in[2];   // [4096]
    float* out = (float*)d_out;

    // Workspace layout: hr (16KB f32) | Ct bf16 (32MB) | optional Abf bf16 (128MB)
    const size_t hr_bytes  = (size_t)NMODE * sizeof(float);
    const size_t ct_bytes  = (size_t)NMODE * NMODE * sizeof(unsigned short);
    const size_t abf_bytes = (size_t)BATCH * NMODE * sizeof(unsigned short);

    float* hr = (float*)d_ws;
    unsigned short* Ct  = (unsigned short*)((char*)d_ws + hr_bytes);
    unsigned short* Abf = (unsigned short*)((char*)d_ws + hr_bytes + ct_bytes);

    (void)in_sizes; (void)n_in; (void)out_size;

    build_hr<<<NMODE / 256, 256, 0, stream>>>(fw, hr);

    const int nvec8 = (NMODE * NMODE) / 8;
    build_circ<<<nvec8 / 256, 256, 0, stream>>>(hr, Ct);

    dim3 grid(NMODE / 128, BATCH / 128);   // (32, 128)
    if (ws_size >= hr_bytes + ct_bytes + abf_bytes) {
        const int avec8 = (int)(((size_t)BATCH * NMODE) / 8);
        convert_A<<<avec8 / 256, 256, 0, stream>>>(X, Abf);
        fourier_gemm<true><<<grid, 256, 0, stream>>>(X, Abf, Ct, lw, out);
    } else {
        fourier_gemm<false><<<grid, 256, 0, stream>>>(X, nullptr, Ct, lw, out);
    }
}
